// TriU_52553219834525
// MI455X (gfx1250) — matrix operand order repaired and checked by reading, not compiled, not run on hardware
//
#include <hip/hip_runtime.h>
#include <hip/hip_bf16.h>

// CDNA5 / gfx1250 wave32 WMMA GEMM for out = tril(W) @ x + b
//   X: (B*C, T) = (262144, 256) f32,  W: (256,256) f32,  b: (256,) f32
// bf16x3 split precision: x=xh+xl, W=Wh+Wl; acc += xh*Wh + xh*Wl + xl*Wh (f32 acc)
// W slabs staged into padded LDS by the Tensor Data Mover (TDM) in one shot.

typedef __attribute__((ext_vector_type(16))) __bf16       v16bf;
typedef __attribute__((ext_vector_type(8)))  __bf16       v8bf;
typedef __attribute__((ext_vector_type(8)))  float        v8f;
typedef __attribute__((ext_vector_type(4)))  unsigned int v4u;
typedef __attribute__((ext_vector_type(8)))  int          v8i;
typedef __attribute__((ext_vector_type(4)))  int          v4i;

#define T_DIM     256
#define LDS_PITCH 264   // 256 + 8 bf16 pad -> row stride 132 dwords (no LDS bank conflicts)

#if defined(__HIP_DEVICE_COMPILE__) && __has_builtin(__builtin_amdgcn_tensor_load_to_lds)
#define USE_TDM 1
#else
#define USE_TDM 0
#endif

// ---------- prep: mask W to lower triangular, split into bf16 hi/lo ----------
__global__ void TriU_prep_w(const float* __restrict__ W,
                            __bf16* __restrict__ whi,
                            __bf16* __restrict__ wlo) {
    const int i = blockIdx.x;     // output row index of W
    const int j = threadIdx.x;    // k index
    float v = (j <= i) ? W[i * T_DIM + j] : 0.0f;
    __bf16 h = (__bf16)v;
    whi[i * T_DIM + j] = h;
    wlo[i * T_DIM + j] = (__bf16)(v - (float)h);
}

// ---------- main GEMM kernel ----------
__global__ __launch_bounds__(256) void TriU_gemm(
    const float*  __restrict__ x,
    const __bf16* __restrict__ whi_g,
    const __bf16* __restrict__ wlo_g,
    const float*  __restrict__ bias,
    float*        __restrict__ out) {

    extern __shared__ __bf16 smem[];
    __bf16* s_hi = smem;                       // [128][LDS_PITCH]
    __bf16* s_lo = smem + 128 * LDS_PITCH;     // [128][LDS_PITCH] (TDM pad lands here)

    const int tid    = threadIdx.x;
    const int lane   = tid & 31;
    const int wave   = tid >> 5;
    const int lane16 = lane & 15;
    const int hiHalf = lane >> 4;              // 0: lanes 0-15, 1: lanes 16-31
    const int mBase  = blockIdx.x * 128 + wave * 16;
    const int row    = mBase + lane16;         // A fragment: M = lane & 15
    const int kb     = hiHalf ? 8 : 0;         // A: lane half K {0-7,16-23} vs {8-15,24-31}

    // ---- preload + split all A fragments for this wave's 16 rows (K = 256) ----
    v16bf ahi[8], alo[8];
    #pragma unroll
    for (int kt = 0; kt < 8; ++kt) {
        const float* xr = x + (size_t)row * T_DIM + kt * 32 + kb;
        float4 f0 = *(const float4*)(xr + 0);    // K = kb+0..3
        float4 f1 = *(const float4*)(xr + 4);    // K = kb+4..7
        float4 f2 = *(const float4*)(xr + 16);   // K = kb+16..19
        float4 f3 = *(const float4*)(xr + 20);   // K = kb+20..23
        float f[16] = {f0.x, f0.y, f0.z, f0.w, f1.x, f1.y, f1.z, f1.w,
                       f2.x, f2.y, f2.z, f2.w, f3.x, f3.y, f3.z, f3.w};
        #pragma unroll
        for (int i = 0; i < 16; ++i) {
            __bf16 h = (__bf16)f[i];
            ahi[kt][i] = h;
            alo[kt][i] = (__bf16)(f[i] - (float)h);
        }
    }

    const int kb2 = hiHalf ? 8 : 0;            // B: lane half K {0-7,16-23} vs {8-15,24-31}

    #pragma unroll 1
    for (int nh = 0; nh < 2; ++nh) {
        __syncthreads();                       // previous half fully consumed

#if USE_TDM
        // ---- TDM: one tensor_load_to_lds moves both 128x256 bf16 slabs ----
        // 3D tile: X=256 elems/row, Y=128 rows, Z=2 planes (W_hi array -> W_lo array,
        // tensor_dim1_stride = 65536 elems = whi/wlo array distance). pad_enable adds
        // 4 DWORDs (16B) after every 128 DWORDs (one 512B row) => LDS pitch 264 bf16.
        if (wave == 0) {
            const unsigned long long ga =
                (unsigned long long)(const void*)(whi_g + (size_t)nh * 128 * T_DIM);
            const unsigned lds = (unsigned)(unsigned long long)(const void*)s_hi;

            v4u g0;
            g0[0] = 1u;                                       // count=1, user descriptor
            g0[1] = lds;                                      // lds_addr
            g0[2] = (unsigned)(ga & 0xFFFFFFFFull);           // global_addr[31:0]
            g0[3] = (unsigned)((ga >> 32) & 0x1FFFFFFull)     // global_addr[56:32]
                    | (2u << 30);                             // type=2 ("image")

            v8i g1;
            g1[0] = (int)((1u << 16)        // data_size=1 (2B elements)
                        | (1u << 20)        // pad_enable
                        | (6u << 22)        // pad_interval: 2^(6+1)=128 DWORDs (one row)
                        | (3u << 25));      // pad_amount: 4 DWORDs = 16B = 8 bf16
            g1[1] = (int)(256u << 16);      // tensor_dim0[15:0]=256 in [31:16]
            g1[2] = (int)(128u << 16);      // tensor_dim0 hi=0 | tensor_dim1[15:0]=128
            g1[3] = (int)(256u << 16);      // tensor_dim1 hi=0 | tile_dim0=256
            g1[4] = (int)(128u | (2u << 16)); // tile_dim1=128, tile_dim2=2
            g1[5] = 256;                    // tensor_dim0_stride[31:0] = 256 elems
            g1[6] = 0;                      // stride0 hi | tensor_dim1_stride[15:0]=0
            g1[7] = 1;                      // tensor_dim1_stride[47:16]: 65536 elems

            v4i g2;
            g2[0] = 2;                      // tensor_dim2 = 2 (Z planes)
            g2[1] = 0; g2[2] = 0; g2[3] = 0;
            v4i g3 = (v4i)0;
            v8i g4 = (v8i)0;                // unused trailing descriptor slot

            __builtin_amdgcn_tensor_load_to_lds(g0, g1, g2, g3, g4, 0);
            __builtin_amdgcn_s_wait_tensorcnt(0);
        }
#else
        // ---- fallback: cooperative VGPR staging of W_hi / W_lo slabs ----
        {
            const uint4* gh = (const uint4*)(whi_g + (size_t)nh * 128 * T_DIM);
            const uint4* gl = (const uint4*)(wlo_g + (size_t)nh * 128 * T_DIM);
            for (int c = tid; c < 128 * 32; c += 256) {     // 16B chunks: 32 per row
                const int r  = c >> 5;
                const int kc = c & 31;
                *(uint4*)&s_hi[r * LDS_PITCH + kc * 8] = gh[c];
                *(uint4*)&s_lo[r * LDS_PITCH + kc * 8] = gl[c];
            }
        }
#endif
        __syncthreads();                       // publish LDS slab to all 8 waves

        for (int nt = 0; nt < 8; ++nt) {
            const int colLocal = nt * 16 + lane16;          // B fragment: N = lane & 15
            const int col      = nh * 128 + colLocal;

            // accumulator init = bias (all 8 acc elements of this lane share `col`)
            const float bv = bias[col];
            v8f c;
            #pragma unroll
            for (int i = 0; i < 8; ++i) c[i] = bv;

            #pragma unroll
            for (int kt = 0; kt < 8; ++kt) {
                const __bf16* ph = &s_hi[colLocal * LDS_PITCH + kt * 32 + kb2];
                const __bf16* pl = &s_lo[colLocal * LDS_PITCH + kt * 32 + kb2];
                v8bf h0 = *(const v8bf*)(ph);
                v8bf h1 = *(const v8bf*)(ph + 16);
                v8bf l0 = *(const v8bf*)(pl);
                v8bf l1 = *(const v8bf*)(pl + 16);
                v16bf bhi, blo;
                #pragma unroll
                for (int i = 0; i < 8; ++i) {
                    bhi[i] = h0[i]; bhi[i + 8] = h1[i];
                    blo[i] = l0[i]; blo[i + 8] = l1[i];
                }
                // bf16x3: xh*Wh + xh*Wl + xl*Wh, f32 accumulate
                c = __builtin_amdgcn_wmma_f32_16x16x32_bf16(false, ahi[kt], false, bhi,
                                                            (short)0, c, false, false);
                c = __builtin_amdgcn_wmma_f32_16x16x32_bf16(false, ahi[kt], false, blo,
                                                            (short)0, c, false, false);
                c = __builtin_amdgcn_wmma_f32_16x16x32_bf16(false, alo[kt], false, bhi,
                                                            (short)0, c, false, false);
            }

            // ---- store 16x16 tile: acc VGPR r -> (M = r + 8*hiHalf, N = lane&15) ----
            float* op = out + (size_t)(mBase + hiHalf * 8) * T_DIM + col;
            #pragma unroll
            for (int r = 0; r < 8; ++r)
                op[(size_t)r * T_DIM] = c[r];
        }
    }
}

extern "C" void kernel_launch(void* const* d_in, const int* in_sizes, int n_in,
                              void* d_out, int out_size, void* d_ws, size_t ws_size,
                              hipStream_t stream) {
    const float* x = (const float*)d_in[0];   // (B, C, T) f32
    const float* W = (const float*)d_in[1];   // (T, T)    f32
    const float* b = (const float*)d_in[2];   // (T,)      f32
    float* out = (float*)d_out;

    __bf16* whi = (__bf16*)d_ws;
    __bf16* wlo = whi + T_DIM * T_DIM;        // contiguous: 65536 elems after whi

    // 1) mask + bf16 hi/lo split of W (tiny, once per launch, deterministic)
    TriU_prep_w<<<T_DIM, T_DIM, 0, stream>>>(W, whi, wlo);

    // 2) main GEMM: each WG = 8 waves covers 128 M-rows x full N=256
    const int M = in_sizes[0] / T_DIM;        // B*C = 262144
    const int grid = M / 128;                 // 2048
    const size_t smem = (size_t)2 * 128 * LDS_PITCH * sizeof(__bf16);  // 132 KB
    TriU_gemm<<<grid, 256, smem, stream>>>(x, whi, wlo, b, out);
}
